// MaskedLSTM_12163347383284
// MI455X (gfx1250) — matrix operand order repaired and checked by reading, not compiled, not run on hardware
//
#include <hip/hip_runtime.h>
#include <hip/hip_bf16.h>

// ---------------------------------------------------------------------------
// MaskedLSTM on MI455X (gfx1250, wave32, WMMA + Tensor Data Mover)
//   B=128, S=1024, I=H=512.  Dead 'i' gate skipped (reference bug reproduced).
//   Phase 1: convert x / W / U to bf16.
//   Phase 2: big GEMM  xc[s][g][b][j] = sigmoid(x@Wm^T) * (x@W^T)   (412 GFLOP)
//   Phase 3: persistent recurrent kernel, 192 WGs, grid barrier per half-step,
//            U/Um weight fragments resident in LDS for all 1024 steps,
//            A-tiles staged global->LDS by the TDM (tensor_load_to_lds),
//            cell state c kept in registers.
// ---------------------------------------------------------------------------

typedef __attribute__((ext_vector_type(16))) __bf16 v16bf;
typedef __attribute__((ext_vector_type(8)))  float  v8f;
typedef __attribute__((ext_vector_type(4)))  unsigned su4;
typedef __attribute__((ext_vector_type(8)))  unsigned su8;

struct alignas(16) U128 { unsigned u0, u1, u2, u3; };
struct alignas(32) U256 { U128 a, b; };

__device__ __forceinline__ v16bf frag_cast(U128 lo, U128 hi) {
    U256 t; t.a = lo; t.b = hi;
    return __builtin_bit_cast(v16bf, t);
}

__device__ __forceinline__ unsigned short f32_to_bf16_rne(float f) {
    unsigned u = __float_as_uint(f);
    unsigned r = u + 0x7FFFu + ((u >> 16) & 1u);
    return (unsigned short)(r >> 16);
}

__device__ __forceinline__ float sigmoidf_fast(float x) {
    return 1.0f / (1.0f + __expf(-x));
}

#define LDA   (512 + 8)   // padded LDS row stride (ushorts) for the A tile
#define NWG   192         // persistent grid size
#define GSIZE 65536       // 128*512 elements per gate plane

// --------------------- TDM descriptor for a 16x512 bf16 tile ----------------
// D# group0: count=1 | lds_addr | global_addr(57b) | type=2.
// D# group1: data_size=2B, pad_enable, pad_interval=256 DW (one 1024B row),
//            pad_amount=4 DW (8-ushort row pad), tensor 512x16, tile 512x16,
//            dim0 stride 512.
__device__ __forceinline__ su4 tdm_g0(unsigned lds_addr, unsigned long long ga) {
    su4 g0;
    g0.x = 1u;                                   // count = 1 valid descriptor
    g0.y = lds_addr;                             // bits 63:32
    g0.z = (unsigned)ga;                         // bits 95:64
    g0.w = (unsigned)(ga >> 32) | (2u << 30);    // addr[56:32] | type=2
    return g0;
}

__device__ __forceinline__ su8 tdm_g1_16x512_padded() {
    su8 g1;
    g1.s0 = (1u << 16)        // data_size = 1 -> 2 bytes
          | (1u << 20)        // pad_enable
          | (7u << 22)        // pad_interval: 256 DWORDs (one 1024B row)
          | (3u << 25);       // pad_amount: 4 DWORDs (16B = 8 ushorts)
    g1.s1 = (512u << 16);     // tensor_dim0[15:0] = 512 @ bits 63:48
    g1.s2 = (16u  << 16);     // tensor_dim0 hi=0 | tensor_dim1[15:0]=16
    g1.s3 = (512u << 16);     // tensor_dim1 hi=0 | tile_dim0=512
    g1.s4 = 16u;              // tile_dim1=16, tile_dim2=0
    g1.s5 = 512u;             // tensor_dim0_stride = 512 (data_size units)
    g1.s6 = 0u;
    g1.s7 = 0u;
    return g1;
}

// Issued once per wave (TDM ignores EXEC; gate by wave-uniform branch).
// readfirstlane every dword so the "s" constraints provably get SGPR tuples
// (without this, LLVM conservatively allocated the runtime-computed group-0
// descriptor to VGPRs and the assembler rejected the operand).
__device__ __forceinline__ void tdm_load_to_lds(su4 g0, su8 g1) {
    su4 s0;
    s0.x = __builtin_amdgcn_readfirstlane(g0.x);
    s0.y = __builtin_amdgcn_readfirstlane(g0.y);
    s0.z = __builtin_amdgcn_readfirstlane(g0.z);
    s0.w = __builtin_amdgcn_readfirstlane(g0.w);
    su8 s1;
    s1.s0 = __builtin_amdgcn_readfirstlane(g1.s0);
    s1.s1 = __builtin_amdgcn_readfirstlane(g1.s1);
    s1.s2 = __builtin_amdgcn_readfirstlane(g1.s2);
    s1.s3 = __builtin_amdgcn_readfirstlane(g1.s3);
    s1.s4 = __builtin_amdgcn_readfirstlane(g1.s4);
    s1.s5 = __builtin_amdgcn_readfirstlane(g1.s5);
    s1.s6 = __builtin_amdgcn_readfirstlane(g1.s6);
    s1.s7 = __builtin_amdgcn_readfirstlane(g1.s7);
    asm volatile("tensor_load_to_lds %0, %1" :: "s"(s0), "s"(s1) : "memory");
}

// -------------------------------- fp32 -> bf16 ------------------------------
__global__ void cvt_f32_bf16(const float* __restrict__ in,
                             unsigned short* __restrict__ out, int n4) {
    int i = blockIdx.x * blockDim.x + threadIdx.x;
    int stride = gridDim.x * blockDim.x;
    for (; i < n4; i += stride) {
        float4 v = ((const float4*)in)[i];
        unsigned lo = (unsigned)f32_to_bf16_rne(v.x) |
                      ((unsigned)f32_to_bf16_rne(v.y) << 16);
        unsigned hi = (unsigned)f32_to_bf16_rne(v.z) |
                      ((unsigned)f32_to_bf16_rne(v.w) << 16);
        ((uint2*)out)[i] = make_uint2(lo, hi);
    }
}

// ------------------- Phase 2: xc = sigmoid(x@Wm^T) * (x@W^T) ----------------
// Grid: rowtile(8192) x gate(3) x jgroup(8); 128 threads (4 waves), wave -> 1 jtile
__global__ __launch_bounds__(128) void xw_gemm(
        const unsigned short* __restrict__ xb,   // [B*S][512] bf16, row = b*1024+s
        const unsigned short* __restrict__ wb,   // [6][512][512] bf16: Wf,Wfm,Wo,Wom,Wc,Wcm
        float* __restrict__ xc)                  // [S][3][B][512]
{
    __shared__ unsigned short aT[16 * LDA];

    int id   = blockIdx.x;
    int rt   = id / 24;
    int rem  = id % 24;
    int g    = rem / 8;
    int jg   = rem % 8;
    int tid  = threadIdx.x;
    int lane = tid & 31;
    int w    = tid >> 5;
    int jt   = jg * 4 + w;
    int n    = lane & 15, sel = lane >> 4;

    // TDM: DMA the 16x512 A tile (rows rt*16..rt*16+15) global -> padded LDS
    if (w == 0) {
        unsigned long long ga = (unsigned long long)(size_t)(xb + (size_t)rt * 16 * 512);
        tdm_load_to_lds(tdm_g0((unsigned)(size_t)&aT[0], ga), tdm_g1_16x512_padded());
        __builtin_amdgcn_s_wait_tensorcnt(0);
    }
    __syncthreads();

    const unsigned short* Wa = wb + (size_t)(2 * g)     * 512 * 512
                                  + (size_t)(jt * 16 + n) * 512 + 16 * sel;
    const unsigned short* Wm = wb + (size_t)(2 * g + 1) * 512 * 512
                                  + (size_t)(jt * 16 + n) * 512 + 16 * sel;
    const unsigned short* arow = &aT[(lane & 15) * LDA + 16 * sel];

    v8f accA = {0,0,0,0,0,0,0,0};
    v8f accM = {0,0,0,0,0,0,0,0};
    #pragma unroll
    for (int ks = 0; ks < 16; ++ks) {
        int k0 = ks * 32;
        v16bf a  = frag_cast(*(const U128*)(arow + k0),
                             *(const U128*)(arow + k0 + 8));
        v16bf bu = frag_cast(*(const U128*)(Wa + k0),
                             *(const U128*)(Wa + k0 + 8));
        v16bf bm = frag_cast(*(const U128*)(Wm + k0),
                             *(const U128*)(Wm + k0 + 8));
        accA = __builtin_amdgcn_wmma_f32_16x16x32_bf16(false, a, false, bu,
                                                       (short)0, accA, false, false);
        accM = __builtin_amdgcn_wmma_f32_16x16x32_bf16(false, a, false, bm,
                                                       (short)0, accM, false, false);
    }

    int col = jt * 16 + n;
    #pragma unroll
    for (int r = 0; r < 8; ++r) {
        int m = r + 8 * sel;
        int R = rt * 16 + m;           // global row = b*1024 + s
        int b = R >> 10, s = R & 1023;
        xc[((size_t)(s * 3 + g) * 128 + b) * 512 + col] =
            sigmoidf_fast(accM[r]) * accA[r];
    }
}

// ------------------------------ grid barrier --------------------------------
__device__ __forceinline__ void grid_barrier(unsigned* cnt, unsigned target) {
    __threadfence();
    __syncthreads();
    if (threadIdx.x == 0) {
        atomicAdd(cnt, 1u);
        while (*(volatile unsigned*)cnt < target)
            __builtin_amdgcn_s_sleep(1);
    }
    __syncthreads();
    __threadfence();
}

// ---------------- Phase 3: persistent recurrent kernel ----------------------
// 192 WGs = gate(3) x btile(8) x jgroup(8); 128 threads, wave -> 1 jtile.
// U/Um fragments live in LDS (frag-order) for the whole sequence.
__global__ __launch_bounds__(128) void lstm_persistent(
        const unsigned short* __restrict__ ub,   // [6][512][512] bf16: Uf,Ufm,Uo,Uom,Uc,Ucm
        const float* __restrict__ xc,            // [S][3][B][512]
        const float* __restrict__ bfv,
        const float* __restrict__ bov,
        const float* __restrict__ bcv,
        float* __restrict__ gates,               // [3][B][512]
        unsigned short* __restrict__ hb,         // [B][512] bf16 (h state)
        float* __restrict__ out,                 // [B][S][512] ++ hT ++ cT
        unsigned* __restrict__ cnt)
{
    extern __shared__ unsigned char smem[];
    unsigned short* aT = (unsigned short*)smem;                  // 16*LDA ushorts
    unsigned short* bF = (unsigned short*)(smem + 16 * LDA * 2); // frag-order B

    int id   = blockIdx.x;
    int g    = id / 64;
    int rem  = id % 64;
    int bt   = rem / 8;
    int jg   = rem % 8;
    int tid  = threadIdx.x;
    int lane = tid & 31;
    int w    = tid >> 5;
    int jt   = jg * 4 + w;
    int n    = lane & 15, sel = lane >> 4;
    int gtid = id * 128 + tid;                   // 0 .. 24575

    // Preload U/Um WMMA fragments for this WG's 4 jtiles into LDS (frag order):
    // entry(slot,mat,ks): 32 lanes x 16 ushorts (one b256 fragment per lane).
    #pragma unroll
    for (int mat = 0; mat < 2; ++mat) {
        const unsigned short* U = ub + ((size_t)(2 * g + mat) * 512
                                        + (size_t)(jt * 16 + n)) * 512 + 16 * sel;
        for (int ks = 0; ks < 16; ++ks) {
            const U128* sp = (const U128*)(U + ks * 32);
            U128 v0 = sp[0], v1 = sp[1];
            U128* dp = (U128*)(bF + ((size_t)((w * 2 + mat) * 16 + ks) * 32 + lane) * 16);
            dp[0] = v0; dp[1] = v1;
        }
    }

    // Init: h = 0 (c lives in registers)
    for (int i = gtid; i < GSIZE; i += NWG * 128) hb[i] = 0;
    float creg[3] = {0.f, 0.f, 0.f};

    // TDM descriptor for this WG's 16x512 h tile (constant across steps)
    su4 hg0 = tdm_g0((unsigned)(size_t)&aT[0],
                     (unsigned long long)(size_t)(hb + (size_t)bt * 16 * 512));
    su8 hg1 = tdm_g1_16x512_padded();

    float bias = (g == 0 ? bfv : (g == 1 ? bov : bcv))[jt * 16 + n];
    int   col  = jt * 16 + n;

    unsigned epoch = 1;
    grid_barrier(cnt, NWG * epoch);              // h init visible everywhere

    for (int s = 0; s < 1024; ++s) {
        // ---- TDM: DMA this WG's 16x512 h tile into LDS ----
        if (w == 0) {
            tdm_load_to_lds(hg0, hg1);
            __builtin_amdgcn_s_wait_tensorcnt(0);
        }
        __syncthreads();

        // ---- GEMM: h @ U^T and h @ Um^T (K = 512) ----
        const unsigned short* arow = &aT[(lane & 15) * LDA + 16 * sel];
        v8f accA = {0,0,0,0,0,0,0,0};
        v8f accM = {0,0,0,0,0,0,0,0};
        #pragma unroll
        for (int ks = 0; ks < 16; ++ks) {
            v16bf a = frag_cast(*(const U128*)(arow + ks * 32),
                                *(const U128*)(arow + ks * 32 + 8));
            const U128* bup = (const U128*)(bF + ((size_t)((w * 2 + 0) * 16 + ks) * 32 + lane) * 16);
            const U128* bmp = (const U128*)(bF + ((size_t)((w * 2 + 1) * 16 + ks) * 32 + lane) * 16);
            v16bf bu = frag_cast(bup[0], bup[1]);
            v16bf bm = frag_cast(bmp[0], bmp[1]);
            accA = __builtin_amdgcn_wmma_f32_16x16x32_bf16(false, a, false, bu,
                                                           (short)0, accA, false, false);
            accM = __builtin_amdgcn_wmma_f32_16x16x32_bf16(false, a, false, bm,
                                                           (short)0, accM, false, false);
        }

        // prefetch next step's xc slice while the stores drain
        if (s + 1 < 1024) {
            const float* nx = xc + (size_t)((s + 1) * 3 + g) * GSIZE;
            __builtin_prefetch(nx + (((rem * 128 + tid) * 32) & (GSIZE - 1)), 0, 1);
        }

        // ---- masked combine: pre = sigmoid(h@Um)* (h@U) + xc + bias ----
        const float* xcs = xc + (size_t)(s * 3 + g) * GSIZE;
        #pragma unroll
        for (int r = 0; r < 8; ++r) {
            int m = r + 8 * sel;
            int b = bt * 16 + m;
            float pre = sigmoidf_fast(accM[r]) * accA[r] + xcs[b * 512 + col] + bias;
            gates[(size_t)(g * 128 + b) * 512 + col] = pre;
        }

        grid_barrier(cnt, NWG * (++epoch));      // gates complete

        // ---- cell update: c = f*c + tanh(gc); h = o*c ----
        int kk = 0;
        for (int i = gtid; i < GSIZE; i += NWG * 128, ++kk) {
            int b = i >> 9, j = i & 511;
            float f = sigmoidf_fast(gates[i]);
            float o = sigmoidf_fast(gates[GSIZE + i]);
            float cn = f * creg[kk] + tanhf(gates[2 * GSIZE + i]);
            creg[kk] = cn;
            float h = o * cn;
            out[((size_t)b * 1024 + s) * 512 + j] = h;
            hb[i] = f32_to_bf16_rne(h);
            if (s == 1023) {                      // hT, cT tails
                out[(size_t)67108864 + i]         = h;
                out[(size_t)67108864 + GSIZE + i] = cn;
            }
        }

        grid_barrier(cnt, NWG * (++epoch));      // h visible for next step
    }
}

// ---------------------------------------------------------------------------
extern "C" void kernel_launch(void* const* d_in, const int* in_sizes, int n_in,
                              void* d_out, int out_size, void* d_ws, size_t ws_size,
                              hipStream_t stream) {
    (void)in_sizes; (void)n_in; (void)out_size; (void)ws_size;

    // input order: Wf,Wf_m,Wi,Wi_m,Wo,Wo_m,Wc,Wc_m, Uf,Uf_m,Ui,Ui_m,Uo,Uo_m,Uc,Uc_m,
    //              bf,bi,bo,bc, x     ('i' gate inputs are dead -> skipped)
    const float* bfv = (const float*)d_in[16];
    const float* bov = (const float*)d_in[18];
    const float* bcv = (const float*)d_in[19];
    const float* x   = (const float*)d_in[20];

    // workspace layout (~944 MB)
    char* ws = (char*)d_ws;
    unsigned*       cnt = (unsigned*)ws;                       size_t off = 256;
    unsigned short* xb  = (unsigned short*)(ws + off);         off += (size_t)67108864 * 2;
    unsigned short* wb  = (unsigned short*)(ws + off);         off += (size_t)6 * 262144 * 2;
    unsigned short* ub  = (unsigned short*)(ws + off);         off += (size_t)6 * 262144 * 2;
    float*          xc  = (float*)(ws + off);                  off += (size_t)1024 * 3 * GSIZE * 4;
    float*          gts = (float*)(ws + off);                  off += (size_t)3 * GSIZE * 4;
    unsigned short* hb  = (unsigned short*)(ws + off);

    (void)hipMemsetAsync(cnt, 0, 256, stream);  // grid-barrier counter must start at 0

    // fp32 -> bf16 conversions
    cvt_f32_bf16<<<4096, 256, 0, stream>>>(x, xb, 67108864 / 4);
    const int wsrc[6] = {0, 1, 4, 5, 6, 7};     // Wf,Wfm,Wo,Wom,Wc,Wcm
    const int usrc[6] = {8, 9, 12, 13, 14, 15}; // Uf,Ufm,Uo,Uom,Uc,Ucm
    for (int i = 0; i < 6; ++i)
        cvt_f32_bf16<<<256, 256, 0, stream>>>((const float*)d_in[wsrc[i]],
                                              wb + (size_t)i * 262144, 262144 / 4);
    for (int i = 0; i < 6; ++i)
        cvt_f32_bf16<<<256, 256, 0, stream>>>((const float*)d_in[usrc[i]],
                                              ub + (size_t)i * 262144, 262144 / 4);

    // Phase 2: all x-side masked-linears as one batched WMMA GEMM
    xw_gemm<<<8192 * 3 * 8, 128, 0, stream>>>(xb, wb, xc);

    // Phase 3: persistent recurrent kernel (192 WGs, ~144 KB LDS each)
    size_t shmem = (size_t)16 * LDA * 2 + (size_t)4 * 2 * 16 * 32 * 16 * 2;
    lstm_persistent<<<NWG, 128, shmem, stream>>>(ub, xc, bfv, bov, bcv,
                                                 gts, hb, (float*)d_out, cnt);
}
